// PraxisMemory_50525995270268
// MI455X (gfx1250) — matrix operand order repaired and checked by reading, not compiled, not run on hardware
//
#include <hip/hip_runtime.h>
#include <hip/hip_bf16.h>

typedef __attribute__((ext_vector_type(16))) __bf16 v16bf;
typedef __attribute__((ext_vector_type(8)))  float  v8f;

constexpr int B_  = 2;
constexpr int H_  = 8;
constexpr int S_  = 2048;
constexpr int D_  = 128;
constexpr int M_  = 32768;
constexpr int NQ  = B_ * S_;     // 4096 queries per head
constexpr int KNN = 16;
constexpr int QT  = 32;          // queries per workgroup (2 WMMA M-tiles)
constexpr int NT  = 128;         // keys per iteration per workgroup (8 waves * 16)
constexpr int NITER = M_ / NT;   // 256

__device__ __forceinline__ unsigned short f2bf(float f) {
  unsigned int u = __float_as_uint(f);
  u += 0x7FFFu + ((u >> 16) & 1u);   // round-to-nearest-even
  return (unsigned short)(u >> 16);
}

union FragAB { v16bf v; uint4 u[2]; };

// ---------------------------------------------------------------------------
// Kernel 1: l2-normalize queries, transpose [B,H,S,D] -> [H, B*S, D], emit bf16
// ---------------------------------------------------------------------------
__global__ void praxis_qnorm_kernel(const float* __restrict__ query,
                                    unsigned short* __restrict__ qn) {
  const int row = blockIdx.x;          // h * NQ + nq
  const int h  = row / NQ;
  const int nq = row % NQ;
  const int b  = nq / S_;
  const int s  = nq % S_;
  const int d  = threadIdx.x;          // 0..127

  const float* src = query + ((((size_t)b * H_ + h) * S_ + s) * D_);
  float v  = src[d];
  float ss = v * v;
  #pragma unroll
  for (int off = 16; off > 0; off >>= 1) ss += __shfl_xor(ss, off, 32);

  __shared__ float red[4];
  const int wid  = threadIdx.x >> 5;
  const int lane = threadIdx.x & 31;
  if (lane == 0) red[wid] = ss;
  __syncthreads();
  const float tot = red[0] + red[1] + red[2] + red[3];
  const float inv = 1.0f / fmaxf(sqrtf(tot), 1e-12f);
  qn[(size_t)row * D_ + d] = f2bf(v * inv);
}

// ---------------------------------------------------------------------------
// Kernel 2: convert key_memories f32 -> bf16 (vectorized 4-wide)
// ---------------------------------------------------------------------------
__global__ void praxis_kconv_kernel(const float* __restrict__ km,
                                    unsigned short* __restrict__ kb) {
  const size_t i = ((size_t)blockIdx.x * blockDim.x + threadIdx.x) * 4;
  float4 f = *(const float4*)(km + i);
  ushort4 o;
  o.x = f2bf(f.x); o.y = f2bf(f.y); o.z = f2bf(f.z); o.w = f2bf(f.w);
  *(ushort4*)(kb + i) = o;
}

// ---------------------------------------------------------------------------
// Kernel 3: WMMA similarity GEMM (2 M-tiles per wave, shared B fragments)
//           + per-query top-16 + gather/blend
// grid = (NQ/QT, H), block = 256 (8 wave32)
// ---------------------------------------------------------------------------
__global__ void __launch_bounds__(256)
praxis_main_kernel(const unsigned short* __restrict__ qn,
                   const unsigned short* __restrict__ kb,
                   const float* __restrict__ vmem,
                   const float* __restrict__ outs,
                   const float* __restrict__ gate,
                   float* __restrict__ out) {
  const int qtile = blockIdx.x;        // 0..127
  const int h     = blockIdx.y;        // 0..7
  const int t     = threadIdx.x;
  const int wave  = t >> 5;
  const int lane  = t & 31;

  __shared__ unsigned short sA[QT * D_];   // 8 KB staged queries (bf16)
  __shared__ float sScores[QT * NT];       // 16 KB score tile / merge scores
  __shared__ int   sMI[QT * NT];           // 16 KB merge indices
  __shared__ float sFS[QT * KNN];          // 2 KB
  __shared__ int   sFI[QT * KNN];          // 2 KB

  // ---- stage 32 query rows (bf16) into LDS: 256 threads x 2 x 16B = 8KB ----
  {
    const uint4* gq = (const uint4*)(qn + ((size_t)h * NQ + (size_t)qtile * QT) * D_);
    ((uint4*)sA)[t]       = gq[t];
    ((uint4*)sA)[t + 256] = gq[t + 256];
  }
  __syncthreads();

  // ---- A fragments for two 16-row tiles, invariant over the key loop ----
  // ISA layout: lanes 0-15 hold M=lane, K {0..7, 16..23}; lanes 16-31 K {8..15, 24..31}
  FragAB a0[4], a1[4];
  {
    const int m  = lane & 15;
    const int kh = (lane >> 4) * 8;
    const unsigned short* ar0 = sA + m * D_;
    const unsigned short* ar1 = sA + (m + 16) * D_;
    #pragma unroll
    for (int kt = 0; kt < 4; ++kt) {
      a0[kt].u[0] = *(const uint4*)(ar0 + kt * 32 + kh);
      a0[kt].u[1] = *(const uint4*)(ar0 + kt * 32 + 16 + kh);
      a1[kt].u[0] = *(const uint4*)(ar1 + kt * 32 + kh);
      a1[kt].u[1] = *(const uint4*)(ar1 + kt * 32 + 16 + kh);
    }
  }

  // ---- per-thread top-16 lists: all 256 threads (32 queries x 8 segments) ----
  float ts[KNN]; int ti[KNN];
  #pragma unroll
  for (int j = 0; j < KNN; ++j) { ts[j] = -__builtin_inff(); ti[j] = 0; }
  const int qrow_sel = t >> 3;   // 0..31
  const int seg      = t & 7;    // 0..7

  const unsigned short* khead = kb + (size_t)h * M_ * D_;

  for (int mt = 0; mt < NITER; ++mt) {
    // ---- each wave: 16 keys, two query tiles -> 8 WMMAs on shared B ----
    const int n_key = mt * NT + wave * 16 + (lane & 15);
    const unsigned short* krow = khead + (size_t)n_key * D_;
    const int kl = (lane >> 4) * 8;    // B layout: lanes 0-15 K {0..7, 16..23}, lanes 16-31 K {8..15, 24..31}
    v8f c0 = {};
    v8f c1 = {};
    #pragma unroll
    for (int kt = 0; kt < 4; ++kt) {
      FragAB bf;
      bf.u[0] = *(const uint4*)(krow + kt * 32 + kl);
      bf.u[1] = *(const uint4*)(krow + kt * 32 + kl + 16);
      c0 = __builtin_amdgcn_wmma_f32_16x16x32_bf16(false, a0[kt].v, false, bf.v,
                                                   (short)0, c0, false, false);
      c1 = __builtin_amdgcn_wmma_f32_16x16x32_bf16(false, a1[kt].v, false, bf.v,
                                                   (short)0, c1, false, false);
    }
    if (mt + 1 < NITER) {
      __builtin_prefetch(krow + (size_t)NT * D_, 0, 3);  // next key tile
    }

    __syncthreads();   // previous iteration's scan finished reading sScores
    {
      // C layout: VGPR v -> M = v + 8*(lane>=16), N = lane&15
      const int mbase = (lane >> 4) * 8;
      const int col   = wave * 16 + (lane & 15);
      #pragma unroll
      for (int v = 0; v < 8; ++v) {
        sScores[(mbase + v) * NT + col]      = c0[v];
        sScores[(mbase + v + 16) * NT + col] = c1[v];
      }
    }
    __syncthreads();

    // ---- all 256 threads: insertion into register-resident sorted top-16 ----
    {
      const int base  = qrow_sel * NT + seg * 16;
      const int gbase = mt * NT + seg * 16;
      #pragma unroll
      for (int j = 0; j < 16; ++j) {
        const float sc = sScores[base + j];
        if (sc > ts[KNN - 1]) {
          ts[KNN - 1] = sc; ti[KNN - 1] = gbase + j;
          #pragma unroll
          for (int p = KNN - 1; p > 0; --p) {
            if (ts[p] > ts[p - 1]) {
              float tf = ts[p]; ts[p] = ts[p - 1]; ts[p - 1] = tf;
              int   tq = ti[p]; ti[p] = ti[p - 1]; ti[p - 1] = tq;
            }
          }
        }
      }
    }
  }
  __syncthreads();

  // ---- dump per-thread partial lists for the 8-way merge ----
  {
    const int base = qrow_sel * NT + seg * 16;
    #pragma unroll
    for (int j = 0; j < KNN; ++j) {
      sScores[base + j] = ts[j];
      sMI[base + j]     = ti[j];
    }
  }
  __syncthreads();

  // ---- 32 threads: select final top-16 per query from 128 candidates ----
  if (t < QT) {
    for (int k = 0; k < KNN; ++k) {
      float best = -__builtin_inff(); int be = 0;
      for (int e = 0; e < NT; ++e) {
        const float v = sScores[t * NT + e];
        if (v > best) { best = v; be = e; }
      }
      sScores[t * NT + be] = -__builtin_inff();
      sFS[t * KNN + k] = best;
      sFI[t * KNN + k] = sMI[t * NT + be];
    }
  }
  __syncthreads();

  // ---- weighted gather of value rows + sigmoid-gate blend (2 queries/thread) ----
  {
    const float g = 1.0f / (1.0f + __expf(-gate[h]));
    const float* vhead = vmem + (size_t)h * M_ * D_;
    #pragma unroll
    for (int pass = 0; pass < 2; ++pass) {
      const int qrow = (t >> 4) + 16 * pass;   // 0..31
      const int dseg = (t & 15) * 8;           // 0..120
      float acc[8];
      #pragma unroll
      for (int j = 0; j < 8; ++j) acc[j] = 0.0f;
      for (int k = 0; k < KNN; ++k) {
        const float  sc   = sFS[qrow * KNN + k];
        const float* vrow = vhead + (size_t)sFI[qrow * KNN + k] * D_ + dseg;
        float4 p0 = *(const float4*)(vrow);
        float4 p1 = *(const float4*)(vrow + 4);
        acc[0] += sc * p0.x; acc[1] += sc * p0.y;
        acc[2] += sc * p0.z; acc[3] += sc * p0.w;
        acc[4] += sc * p1.x; acc[5] += sc * p1.y;
        acc[6] += sc * p1.z; acc[7] += sc * p1.w;
      }
      const int nq = qtile * QT + qrow;
      const int b  = nq / S_;
      const int s  = nq % S_;
      const size_t obase = (((size_t)b * H_ + h) * S_ + s) * D_ + dseg;
      #pragma unroll
      for (int j = 0; j < 8; ++j)
        out[obase + j] = g * acc[j] + (1.0f - g) * outs[obase + j];
    }
  }
}

// ---------------------------------------------------------------------------
// Host-side launch
// ---------------------------------------------------------------------------
extern "C" void kernel_launch(void* const* d_in, const int* in_sizes, int n_in,
                              void* d_out, int out_size, void* d_ws, size_t ws_size,
                              hipStream_t stream) {
  // setup_inputs order:
  // 0 inputs, 1 query, 2 key, 3 value, 4 outputs, 5 gate, 6 key_memories, 7 value_memories
  const float* query = (const float*)d_in[1];
  const float* outs  = (const float*)d_in[4];
  const float* gate  = (const float*)d_in[5];
  const float* km    = (const float*)d_in[6];
  const float* vm    = (const float*)d_in[7];
  float* out = (float*)d_out;

  // workspace: bf16 keys (67,108,864 B) then bf16 normalized queries (8,388,608 B)
  unsigned short* keys_bf = (unsigned short*)d_ws;
  unsigned short* qn_bf   = (unsigned short*)((char*)d_ws + (size_t)H_ * M_ * D_ * 2);

  praxis_qnorm_kernel<<<H_ * NQ, 128, 0, stream>>>(query, qn_bf);

  const int kconv_groups = (H_ * M_ * D_) / 4;       // 8,388,608 float4 groups
  praxis_kconv_kernel<<<kconv_groups / 256, 256, 0, stream>>>(km, keys_bf);

  dim3 grid(NQ / QT, H_);
  praxis_main_kernel<<<grid, 256, 0, stream>>>(qn_bf, keys_bf, vm, outs, gate, out);
}
